// Model_25598005084714
// MI455X (gfx1250) — matrix operand order repaired and checked by reading, not compiled, not run on hardware
//
#include <hip/hip_runtime.h>
#include <hip/hip_bf16.h>

typedef __attribute__((ext_vector_type(16))) _Float16 v16h;
typedef __attribute__((ext_vector_type(8)))  _Float16 v8h;
typedef __attribute__((ext_vector_type(8)))  float    v8f;

constexpr int B_   = 256;
constexpr int T_   = 512;
constexpr int H_   = 64;    // hidden
constexpr int NG_  = 256;   // 4*H gates
constexpr int LDSS = 72;    // halves per LDS row (144B, 16B-aligned rows)

// ---- branch-free activations on the trans units (critical-path ops) -------
#define LOG2E_F 1.4426950408889634f

__device__ __forceinline__ float fast_sigmoid(float x) {
    // 1/(1+e^-x) = rcp(1 + exp2(-log2e*x)); saturates correctly at both ends.
    float e = __builtin_amdgcn_exp2f(-LOG2E_F * x);
    return __builtin_amdgcn_rcpf(1.0f + e);
}

__device__ __forceinline__ float fast_tanh(float x) {
#if __has_builtin(__builtin_amdgcn_tanhf)
    return __builtin_amdgcn_tanhf(x);        // native v_tanh_f32 on gfx1250
#else
    // 1 - 2/(e^{2x}+1); exp2->inf => +1, exp2->0 => -1 (branch-free, exact sat)
    float e = __builtin_amdgcn_exp2f(2.0f * LOG2E_F * x);
    return 1.0f - 2.0f * __builtin_amdgcn_rcpf(e + 1.0f);
#endif
}

// ---------------------------------------------------------------------------
// Prep: transpose+convert weights to f16 [K=64][N=256] (k-major so the WMMA
// B-fragment load is a per-lane column gather), fold the two biases together.
// Layer 0 has Kin=2; pad K with zeros so every layer uses the same scan.
// ---------------------------------------------------------------------------
__global__ void prep_weights(const float* __restrict__ W_ih,
                             const float* __restrict__ W_hh,
                             const float* __restrict__ b_ih,
                             const float* __restrict__ b_hh,
                             _Float16* __restrict__ Wt_ih,
                             _Float16* __restrict__ Wt_hh,
                             float* __restrict__ bias, int Kin) {
    int n = blockIdx.x * blockDim.x + threadIdx.x;   // gate output index 0..255
    if (n >= NG_) return;
    for (int k = 0; k < H_; ++k) {
        float wih = (k < Kin) ? W_ih[n * Kin + k] : 0.0f;
        Wt_ih[k * NG_ + n] = (_Float16)wih;
        Wt_hh[k * NG_ + n] = (_Float16)W_hh[n * H_ + k];
    }
    bias[n] = b_ih[n] + b_hh[n];
}

// x [B,T,2] f32 -> padded f16 sequence [B,T,64] so layer 0 reuses the scan.
__global__ void prep_x(const float* __restrict__ x, _Float16* __restrict__ h0,
                       int total) {
    int idx = blockIdx.x * blockDim.x + threadIdx.x;  // token
    if (idx >= total) return;
    _Float16* o = h0 + (size_t)idx * H_;
    o[0] = (_Float16)x[idx * 2 + 0];
    o[1] = (_Float16)x[idx * 2 + 1];
    #pragma unroll
    for (int k = 2; k < H_; ++k) o[k] = (_Float16)0;
}

// ---------------------------------------------------------------------------
// Fused LSTM layer scan. Grid = B/16 workgroups x 128 threads (4 waves).
// Wave w owns hidden cols [16w,16w+16) of all four gates; cell state in VGPRs.
// g_t = x_t @ W_ih^T + h_{t-1} @ W_hh^T + bias   via v_wmma_f32_16x16x32_f16.
// x fragments are software-pipelined one step ahead (off the recurrence chain)
// so their global-load latency hides under compute + barrier.
// ---------------------------------------------------------------------------
__global__ __launch_bounds__(128) void lstm_scan(
    const _Float16* __restrict__ in_seq,   // [B,T,64] f16
    _Float16* __restrict__ out_seq,        // [B,T,64] f16
    const _Float16* __restrict__ Wt_ih,    // [64][256] k-major
    const _Float16* __restrict__ Wt_hh,    // [64][256] k-major
    const float* __restrict__ bias,        // [256]
    int T) {
    __shared__ _Float16 hbuf[2][16 * LDSS];

    const int tid     = threadIdx.x;
    const int wave    = tid >> 5;    // 0..3
    const int lane    = tid & 31;
    const int half_id = lane >> 4;   // lane group
    const int ln      = lane & 15;
    const int row0    = blockIdx.x * 16;   // batch rows of this WG
    const int ntile0  = wave * 16;         // hidden col base for this wave

    // zero both LDS h buffers (h_{-1} = 0)
    for (int i = tid; i < 2 * 16 * LDSS; i += 128)
        ((_Float16*)hbuf)[i] = (_Float16)0;

    // ---- hoist weight B-fragments: 4 gate tiles x 2 K-chunks x {ih,hh} ----
    // Dense-B layout: lane = column n (within tile), element e -> k = e<8 ? hid*8+e : 16+hid*8+(e-8).
    v16h Bih[4][2], Bhh[4][2];
    #pragma unroll
    for (int g = 0; g < 4; ++g) {
        const int nbase = g * H_ + ntile0;   // gate-major column base
        #pragma unroll
        for (int kc = 0; kc < 2; ++kc) {
            v16h bi, bh;
            #pragma unroll
            for (int e = 0; e < 16; ++e) {
                int k = kc * 32 + ((e < 8) ? (half_id * 8 + e) : (16 + half_id * 8 + (e - 8)));
                bi[e] = Wt_ih[k * NG_ + nbase + ln];
                bh[e] = Wt_hh[k * NG_ + nbase + ln];
            }
            Bih[g][kc] = bi;
            Bhh[g][kc] = bh;
        }
    }
    float bval[4];
    #pragma unroll
    for (int g = 0; g < 4; ++g) bval[g] = bias[g * H_ + ntile0 + ln];

    v8f c;                                   // cell state, D-layout
    #pragma unroll
    for (int i = 0; i < 8; ++i) c[i] = 0.0f;

    // A-fragment row pointer: lane = row m (both lane halves map m=0..15)
    const _Float16* xrow = in_seq + (size_t)(row0 + ln) * T * H_;

    // helper to load one step's x A-fragments (4 x 16B per lane)
    auto load_x = [&](int t, v16h xa[2]) {
        const _Float16* xp = xrow + (size_t)t * H_;
        #pragma unroll
        for (int kc = 0; kc < 2; ++kc) {
            v8h lo = *(const v8h*)(xp + kc * 32 + half_id * 8);
            v8h hi = *(const v8h*)(xp + kc * 32 + 16 + half_id * 8);
            v16h a;
            #pragma unroll
            for (int e = 0; e < 8; ++e) { a[e] = lo[e]; a[e + 8] = hi[e]; }
            xa[kc] = a;
        }
    };

    // prologue: prefetch x(0)
    v16h xa_cur[2];
    load_x(0, xa_cur);

    __syncthreads();
    int p = 0;
    for (int t = 0; t < T; ++t) {
        // ---- issue prefetch of x(t+1) immediately (consumed next iter) ----
        v16h xa_nxt[2];
        int tn = t + 1 < T ? t + 1 : T - 1;       // branch-free clamp
        load_x(tn, xa_nxt);

        // ---- h A-fragments from LDS buffer p (A layout: m=lane, k=elems) ----
        v16h ha[2];
        #pragma unroll
        for (int kc = 0; kc < 2; ++kc) {
            #pragma unroll
            for (int e = 0; e < 16; ++e) {
                int klocal = (e < 8) ? (half_id * 8 + e)
                                     : (16 + half_id * 8 + (e - 8));
                ha[kc][e] = hbuf[p][ln * LDSS + kc * 32 + klocal];
            }
        }
        // ---- gate pre-activations: 16 WMMAs (8 off-chain + 8 on-chain) ----
        v8f acc[4];
        #pragma unroll
        for (int g = 0; g < 4; ++g) {
            v8f a;
            #pragma unroll
            for (int i = 0; i < 8; ++i) a[i] = bval[g];
            a = __builtin_amdgcn_wmma_f32_16x16x32_f16(false, xa_cur[0], false, Bih[g][0], (short)0, a, false, false);
            a = __builtin_amdgcn_wmma_f32_16x16x32_f16(false, xa_cur[1], false, Bih[g][1], (short)0, a, false, false);
            a = __builtin_amdgcn_wmma_f32_16x16x32_f16(false, ha[0],     false, Bhh[g][0], (short)0, a, false, false);
            a = __builtin_amdgcn_wmma_f32_16x16x32_f16(false, ha[1],     false, Bhh[g][1], (short)0, a, false, false);
            acc[g] = a;
        }
        // ---- elementwise LSTM cell update (branch-free trans ops) ----
        v8f hnew;
        #pragma unroll
        for (int i = 0; i < 8; ++i) {
            float iv = fast_sigmoid(acc[0][i]);
            float fv = fast_sigmoid(acc[1][i]);
            float gv = fast_tanh(acc[2][i]);
            float ov = fast_sigmoid(acc[3][i]);
            float cv = __builtin_fmaf(fv, c[i], iv * gv);
            c[i] = cv;
            hnew[i] = ov * fast_tanh(cv);
        }
        // ---- D-layout -> row-major f16 into LDS buffer q ----
        const int q = p ^ 1;
        #pragma unroll
        for (int i = 0; i < 8; ++i) {
            int m = half_id * 8 + i;                 // batch row in tile
            hbuf[q][m * LDSS + ntile0 + ln] = (_Float16)hnew[i];
        }
        __syncthreads();
        // ---- coalesced global h store (16B per thread) ----
        {
            int r = tid >> 3;                        // row 0..15
            int cc = (tid & 7) * 8;                  // col chunk
            float4 v = *(const float4*)&hbuf[q][r * LDSS + cc];
            *(float4*)(out_seq + (size_t)(row0 + r) * T * H_ +
                       (size_t)t * H_ + cc) = v;
        }
        xa_cur[0] = xa_nxt[0];
        xa_cur[1] = xa_nxt[1];
        p = q;
    }
}

// out[b,t] = h4[b,t,:] . W_lin + b_lin
__global__ void final_linear(const _Float16* __restrict__ h,
                             const float* __restrict__ Wl,
                             const float* __restrict__ bl,
                             float* __restrict__ out, int total) {
    int idx = blockIdx.x * blockDim.x + threadIdx.x;
    if (idx >= total) return;
    const _Float16* hp = h + (size_t)idx * H_;
    float acc = bl[0];
    #pragma unroll
    for (int j = 0; j < H_; ++j) acc += (float)hp[j] * Wl[j];
    out[idx] = acc;
}

extern "C" void kernel_launch(void* const* d_in, const int* in_sizes, int n_in,
                              void* d_out, int out_size, void* d_ws, size_t ws_size,
                              hipStream_t stream) {
    const float* x = (const float*)d_in[0];
    const float *W_ih[5], *W_hh[5], *b_ih[5], *b_hh[5];
    for (int l = 0; l < 5; ++l) {
        W_ih[l] = (const float*)d_in[1 + 4 * l + 0];
        W_hh[l] = (const float*)d_in[1 + 4 * l + 1];
        b_ih[l] = (const float*)d_in[1 + 4 * l + 2];
        b_hh[l] = (const float*)d_in[1 + 4 * l + 3];
    }
    const float* W_lin = (const float*)d_in[21];
    const float* b_lin = (const float*)d_in[22];
    float* out = (float*)d_out;

    // workspace: two f16 h-sequence buffers (ping-pong) + per-layer weights
    char* ws = (char*)d_ws;
    const size_t hbytes = (size_t)B_ * T_ * H_ * sizeof(_Float16);  // 16.8 MB
    _Float16* hA = (_Float16*)ws;
    _Float16* hB = (_Float16*)(ws + hbytes);
    char* wp = ws + 2 * hbytes;
    _Float16* Wt_ih[5]; _Float16* Wt_hh[5]; float* bias[5];
    for (int l = 0; l < 5; ++l) {
        Wt_ih[l] = (_Float16*)wp; wp += (size_t)H_ * NG_ * sizeof(_Float16);
        Wt_hh[l] = (_Float16*)wp; wp += (size_t)H_ * NG_ * sizeof(_Float16);
        bias[l]  = (float*)wp;    wp += (size_t)NG_ * sizeof(float);
    }

    const int tokens = B_ * T_;
    prep_x<<<(tokens + 255) / 256, 256, 0, stream>>>(x, hA, tokens);
    for (int l = 0; l < 5; ++l)
        prep_weights<<<1, 256, 0, stream>>>(W_ih[l], W_hh[l], b_ih[l], b_hh[l],
                                            Wt_ih[l], Wt_hh[l], bias[l],
                                            l == 0 ? 2 : H_);
    for (int l = 0; l < 5; ++l) {
        const _Float16* in = (l & 1) ? hB : hA;
        _Float16* ol       = (l & 1) ? hA : hB;
        lstm_scan<<<B_ / 16, 128, 0, stream>>>(in, ol, Wt_ih[l], Wt_hh[l],
                                               bias[l], T_);
    }
    final_linear<<<(tokens + 255) / 256, 256, 0, stream>>>(hB, W_lin, b_lin,
                                                           out, tokens);
}
